// EnhancedHyperbolicAttention_16870631539335
// MI455X (gfx1250) — matrix operand order repaired and checked by reading, not compiled, not run on hardware
//
#include <hip/hip_runtime.h>
#include <hip/hip_bf16.h>
#include <math.h>

// ---------------- problem constants (from reference) ----------------
constexpr int kB  = 2;
constexpr int kH  = 16;
constexpr int kN  = 2048;
constexpr int kD  = 1024;
constexpr int kDh = 64;
constexpr int kBN = kB * kN;   // 4096 rows of the big GEMMs
constexpr int kBH = kB * kH;   // 32 attention planes

// ---------------- WMMA types (gfx1250, wave32) ----------------
typedef __attribute__((ext_vector_type(16))) __bf16 v16bf;
typedef __attribute__((ext_vector_type(8)))  float  v8f;

union FragBF {
  v16bf v;
  unsigned short u[16];
  uint4 q[2];
};

__device__ __forceinline__ unsigned short f2bf(float f) {
  unsigned int u = __float_as_uint(f);
  u += 0x7fffu + ((u >> 16) & 1u);        // round-to-nearest-even
  return (unsigned short)(u >> 16);
}
__device__ __forceinline__ float bf2f(unsigned short h) {
  return __uint_as_float(((unsigned int)h) << 16);
}

__device__ __forceinline__ v8f zero8() {
  v8f z = {0.f, 0.f, 0.f, 0.f, 0.f, 0.f, 0.f, 0.f};
  return z;
}

// fast hardware transcendentals (raw v_sqrt/v_log/v_exp/v_rcp, no fixup chains)
__device__ __forceinline__ float fast_sqrt(float x) { return __builtin_amdgcn_sqrtf(x); }
__device__ __forceinline__ float fast_ln(float x)   { return 0.69314718f * __builtin_amdgcn_logf(x); }
__device__ __forceinline__ float fast_exp(float x)  { return __builtin_amdgcn_exp2f(1.44269504f * x); }
__device__ __forceinline__ float fast_rcp(float x)  { return __builtin_amdgcn_rcpf(x); }

// A-frag (16x32 bf16, M x K): lane holds row m = lane&15.
// element e<8  -> K = kb + 8*(lane>>4) + e
// element e>=8 -> K = kb + 16 + 8*(lane>>4) + (e-8)
// src row-major [rows][ld] of bf16 bits. Two 16B loads per lane.
__device__ __forceinline__ v16bf load_A(const unsigned short* __restrict__ base,
                                        int ld, int kb, int lane) {
  int m = lane & 15, hi = lane >> 4;
  FragBF f;
  const unsigned short* p = base + m * ld + kb + 8 * hi;
  f.q[0] = *(const uint4*)p;          // K: kb+8hi .. +7
  f.q[1] = *(const uint4*)(p + 16);   // K: kb+16+8hi .. +7
  return f.v;
}

// B-frag (32x16 bf16, K x N) taken from a ROW-MAJOR [n][K] source, i.e. B = srcᵀ.
// lane holds column n = lane&15; element e<8 -> K = kb + 8*(lane>>4) + e,
// element e>=8 -> K = kb + 16 + 8*(lane>>4) + (e-8). Two 16B reads per lane.
__device__ __forceinline__ v16bf load_Bt(const unsigned short* __restrict__ base,
                                         int ld, int kb, int lane) {
  int n = lane & 15, hi = lane >> 4;
  FragBF f;
  const unsigned short* p = base + n * ld + kb + 8 * hi;
  f.q[0] = *(const uint4*)p;
  f.q[1] = *(const uint4*)(p + 16);
  return f.v;
}

__device__ __forceinline__ v8f wmma_bf16(v16bf a, v16bf b, v8f c) {
  return __builtin_amdgcn_wmma_f32_16x16x32_bf16(false, a, false, b,
                                                 (short)0, c, false, false);
}

// ---------------- kernel 1: f32 -> bf16 bits ----------------
__global__ void f32_to_bf16(const float* __restrict__ src,
                            unsigned short* __restrict__ dst, int n) {
  int i = blockIdx.x * blockDim.x + threadIdx.x;
  if (i < n) dst[i] = f2bf(src[i]);
}

// ---------------- kernel 2: GEMM  y = A @ Wᵀ ----------------
// A: [M,K] bf16 row-major, W: [Nc,K] bf16 row-major. Wave tile: 32(M) x 64(N).
// MODE 0: store f32 to dst[M][Nc]                  (output projection -> d_out)
// MODE 1: store bf16 into [B,H,N,Dh] head layout   (Q, K)
// MODE 2: store bf16 into [B,H,Dh,N] transposed    (V -> contiguous B-frags)
template <int MODE>
__global__ __launch_bounds__(256) void gemm_bf16(
    const unsigned short* __restrict__ A, const unsigned short* __restrict__ W,
    void* __restrict__ dst, int M, int K, int Nc) {
  int lane = threadIdx.x & 31;
  int wave = (blockIdx.x * blockDim.x + threadIdx.x) >> 5;
  int ngroups = Nc >> 6;                 // 64-wide N groups per wave
  int mtile = wave / ngroups;            // 32-row M tiles
  int ng = wave % ngroups;
  if (mtile * 32 >= M) return;

  const unsigned short* Arow0 = A + (size_t)mtile * 32 * K;
  const unsigned short* Arow1 = Arow0 + (size_t)16 * K;
  v8f acc[2][4];
#pragma unroll
  for (int rb = 0; rb < 2; ++rb)
#pragma unroll
    for (int t = 0; t < 4; ++t) acc[rb][t] = zero8();

  for (int kb = 0; kb < K; kb += 32) {
    v16bf a0 = load_A(Arow0, K, kb, lane);
    v16bf a1 = load_A(Arow1, K, kb, lane);
#pragma unroll
    for (int t = 0; t < 4; ++t) {
      v16bf b = load_Bt(W + (size_t)(ng * 64 + 16 * t) * K, K, kb, lane);
      acc[0][t] = wmma_bf16(a0, b, acc[0][t]);
      acc[1][t] = wmma_bf16(a1, b, acc[1][t]);
    }
  }

  int hi = lane >> 4, nl = lane & 15;
#pragma unroll
  for (int rb = 0; rb < 2; ++rb) {
#pragma unroll
    for (int t = 0; t < 4; ++t) {
#pragma unroll
      for (int r = 0; r < 8; ++r) {
        int row = mtile * 32 + rb * 16 + r + 8 * hi;  // C layout
        int col = ng * 64 + 16 * t + nl;
        float val = acc[rb][t][r];
        if (MODE == 0) {
          ((float*)dst)[(size_t)row * Nc + col] = val;
        } else {
          int b = row >> 11;                     // row / kN
          int n = row & (kN - 1);
          int h = col >> 6;
          int d = col & (kDh - 1);
          if (MODE == 1) {
            ((unsigned short*)dst)[(((size_t)(b * kH + h) * kN + n) * kDh) + d] =
                f2bf(val);
          } else {  // MODE 2: [B,H,Dh,N]
            ((unsigned short*)dst)[(((size_t)(b * kH + h) * kDh + d) * kN) + n] =
                f2bf(val);
          }
        }
      }
    }
  }
}

// ---------------- kernel 3: row norms of Q and K ----------------
__global__ void row_norms(const unsigned short* __restrict__ Qb,
                          const unsigned short* __restrict__ Kb,
                          float* __restrict__ qn, float* __restrict__ kn,
                          int rows) {
  int r = blockIdx.x * blockDim.x + threadIdx.x;
  if (r >= rows) return;
  const unsigned short* q = Qb + (size_t)r * kDh;
  const unsigned short* k = Kb + (size_t)r * kDh;
  float sq = 0.f, sk = 0.f;
#pragma unroll
  for (int d = 0; d < kDh; ++d) {
    float a = bf2f(q[d]);
    sq = fmaf(a, a, sq);
    float b = bf2f(k[d]);
    sk = fmaf(b, b, sk);
  }
  qn[r] = sq;
  kn[r] = sk;
}

// hyperbolic distance -> score (matches reference piecewise form)
__device__ __forceinline__ float hyp_score(float qk, float qnv, float knv,
                                           float c, float beta) {
  const float EPS = 1e-8f;
  float diff = fmaxf(qnv + knv - 2.0f * qk, 0.0f);
  float ed = fast_sqrt(diff + EPS);
  float ns = qnv + knv;
  float cn = c * ns;
  float d_taylor = ed * (1.0f + 0.5f * cn + 0.125f * cn * cn);
  float d_asymp = 0.693f + fast_ln(ed + EPS) + 0.25f * cn;
  float d_std = ed * fast_sqrt(1.0f + cn);
  float dist = (ed < 0.1f) ? d_taylor : ((ed > 2.0f) ? d_asymp : d_std);
  return -beta * dist;
}

// ---------------- kernel 4: causal hyperbolic flash attention ----------------
// one wave handles 16 query rows of one (b,h) plane; 32 keys per step.
// V is pre-transposed ([B,H,Dh,N]) so every WMMA operand is a contiguous
// 32-byte-per-lane load (no scalar gathers in the hot loop).
__global__ __launch_bounds__(256) void hyp_attn(
    const unsigned short* __restrict__ Qb, const unsigned short* __restrict__ Kb,
    const unsigned short* __restrict__ Vt, const float* __restrict__ qn,
    const float* __restrict__ kn, const float* __restrict__ log_c,
    const float* __restrict__ log_beta, unsigned short* __restrict__ Ob) {
  __shared__ unsigned short smem[8 * 16 * 32];  // 1KB per wave: P staging
  int lane = threadIdx.x & 31;
  int wIn = threadIdx.x >> 5;
  int wave = blockIdx.x * 8 + wIn;
  int qt = wave & ((kN / 16) - 1);  // query tile 0..127
  int bh = wave >> 7;               // 0..31
  int hi = lane >> 4, nl = lane & 15;

  float lc = log_c[0], lb = log_beta[0];
  float c = (lc > 20.f) ? lc : log1pf(__expf(lc));          // softplus (once)
  float beta = ((lb > 20.f) ? lb : log1pf(__expf(lb))) + 0.5f;

  const unsigned short* Qp = Qb + ((size_t)bh * kN + qt * 16) * kDh;
  const unsigned short* Kp = Kb + (size_t)bh * kN * kDh;
  const unsigned short* Vtp = Vt + (size_t)bh * kDh * kN;   // [Dh][N]
  const float* qnp = qn + (size_t)bh * kN + qt * 16;
  const float* knp = kn + (size_t)bh * kN;

  // Q A-frags for the whole tile (K = 0..31, 32..63)
  v16bf qa0 = load_A(Qp, kDh, 0, lane);
  v16bf qa1 = load_A(Qp, kDh, 32, lane);

  float qrow[8];
#pragma unroll
  for (int r = 0; r < 8; ++r) qrow[r] = qnp[r + 8 * hi];

  v8f o0 = zero8(), o1 = zero8(), o2 = zero8(), o3 = zero8();
  float Mr[8], Lr[8];
#pragma unroll
  for (int r = 0; r < 8; ++r) { Mr[r] = -1e30f; Lr[r] = 0.f; }

  unsigned short* pls = smem + wIn * 512;  // this wave's 16x32 P tile

  int nblk = (qt >> 1) + 1;  // 32-key blocks covering keys <= last query
  for (int blk = 0; blk < nblk; ++blk) {
    int k0 = blk * 32;
    // ---- S = Q Kᵀ  (two 16x16 score tiles) ----
    v8f s0 = zero8(), s1 = zero8();
    {
      v16bf b0 = load_Bt(Kp + (size_t)k0 * kDh, kDh, 0, lane);
      v16bf b1 = load_Bt(Kp + (size_t)k0 * kDh, kDh, 32, lane);
      s0 = wmma_bf16(qa0, b0, s0);
      s0 = wmma_bf16(qa1, b1, s0);
      v16bf b2 = load_Bt(Kp + (size_t)(k0 + 16) * kDh, kDh, 0, lane);
      v16bf b3 = load_Bt(Kp + (size_t)(k0 + 16) * kDh, kDh, 32, lane);
      s1 = wmma_bf16(qa0, b2, s1);
      s1 = wmma_bf16(qa1, b3, s1);
    }
    float kn0 = knp[k0 + nl];
    float kn1 = knp[k0 + 16 + nl];
    int nc0 = k0 + nl, nc1 = k0 + 16 + nl;
    // ---- distance transform + causal mask + online softmax (per row r) ----
#pragma unroll
    for (int r = 0; r < 8; ++r) {
      int mrow = qt * 16 + r + 8 * hi;
      float sc0 = hyp_score(s0[r], qrow[r], kn0, c, beta);
      float sc1 = hyp_score(s1[r], qrow[r], kn1, c, beta);
      if (nc0 > mrow) sc0 = -1e30f;
      if (nc1 > mrow) sc1 = -1e30f;
      float mx = fmaxf(sc0, sc1);
#pragma unroll
      for (int m = 1; m < 16; m <<= 1) mx = fmaxf(mx, __shfl_xor(mx, m, 32));
      float mnew = fmaxf(Mr[r], mx);
      float alpha = fast_exp(Mr[r] - mnew);
      float p0 = fast_exp(sc0 - mnew);
      float p1 = fast_exp(sc1 - mnew);
      float rs = p0 + p1;
#pragma unroll
      for (int m = 1; m < 16; m <<= 1) rs += __shfl_xor(rs, m, 32);
      Lr[r] = Lr[r] * alpha + rs;
      Mr[r] = mnew;
      o0[r] *= alpha; o1[r] *= alpha; o2[r] *= alpha; o3[r] *= alpha;
      // stash P (C layout -> row-major LDS) for A-frag reload
      int mloc = r + 8 * hi;
      pls[mloc * 32 + nl] = f2bf(p0);
      pls[mloc * 32 + 16 + nl] = f2bf(p1);
    }
    asm volatile("s_wait_dscnt 0" ::: "memory");  // LDS RAW fence (CDNA5 split counters)
    // ---- O += P V : B-frags from transposed V, contiguous per lane ----
    v16bf pa = load_A(pls, 32, 0, lane);
    o0 = wmma_bf16(pa, load_Bt(Vtp + (size_t)0 * kN,  kN, k0, lane), o0);
    o1 = wmma_bf16(pa, load_Bt(Vtp + (size_t)16 * kN, kN, k0, lane), o1);
    o2 = wmma_bf16(pa, load_Bt(Vtp + (size_t)32 * kN, kN, k0, lane), o2);
    o3 = wmma_bf16(pa, load_Bt(Vtp + (size_t)48 * kN, kN, k0, lane), o3);
  }

  // ---- normalize and store into [B, N, H*Dh] bf16 for the output GEMM ----
  int b = bh >> 4, h = bh & 15;
#pragma unroll
  for (int r = 0; r < 8; ++r) {
    float inv = fast_rcp(Lr[r]);
    int row = qt * 16 + r + 8 * hi;
    size_t base = ((size_t)b * kN + row) * kD + h * kDh + nl;
    Ob[base + 0]  = f2bf(o0[r] * inv);
    Ob[base + 16] = f2bf(o1[r] * inv);
    Ob[base + 32] = f2bf(o2[r] * inv);
    Ob[base + 48] = f2bf(o3[r] * inv);
  }
}

// ---------------- host-side orchestration ----------------
extern "C" void kernel_launch(void* const* d_in, const int* in_sizes, int n_in,
                              void* d_out, int out_size, void* d_ws,
                              size_t ws_size, hipStream_t stream) {
  (void)in_sizes; (void)n_in; (void)out_size; (void)ws_size;
  const float* x  = (const float*)d_in[0];
  const float* Wq = (const float*)d_in[1];
  const float* Wk = (const float*)d_in[2];
  const float* Wv = (const float*)d_in[3];
  const float* Wo = (const float*)d_in[4];
  const float* log_c = (const float*)d_in[5];
  const float* log_beta = (const float*)d_in[6];

  char* ws = (char*)d_ws;
  size_t off = 0;
  unsigned short* xb  = (unsigned short*)(ws + off); off += (size_t)kBN * kD * 2;   // 8MB
  unsigned short* wqb = (unsigned short*)(ws + off); off += (size_t)kD * kD * 2;    // 2MB
  unsigned short* wkb = (unsigned short*)(ws + off); off += (size_t)kD * kD * 2;
  unsigned short* wvb = (unsigned short*)(ws + off); off += (size_t)kD * kD * 2;
  unsigned short* wob = (unsigned short*)(ws + off); off += (size_t)kD * kD * 2;
  unsigned short* Qb  = (unsigned short*)(ws + off); off += (size_t)kBH * kN * kDh * 2; // 8MB
  unsigned short* Kb  = (unsigned short*)(ws + off); off += (size_t)kBH * kN * kDh * 2;
  unsigned short* Vt  = (unsigned short*)(ws + off); off += (size_t)kBH * kDh * kN * 2;
  float* qn = (float*)(ws + off); off += (size_t)kBH * kN * 4;
  float* kn = (float*)(ws + off); off += (size_t)kBH * kN * 4;
  unsigned short* Ob = (unsigned short*)(ws + off); off += (size_t)kBN * kD * 2;    // 8MB

  // 1) convert inputs to bf16
  {
    int n = kBN * kD;
    f32_to_bf16<<<(n + 255) / 256, 256, 0, stream>>>(x, xb, n);
    int nw = kD * kD;
    f32_to_bf16<<<(nw + 255) / 256, 256, 0, stream>>>(Wq, wqb, nw);
    f32_to_bf16<<<(nw + 255) / 256, 256, 0, stream>>>(Wk, wkb, nw);
    f32_to_bf16<<<(nw + 255) / 256, 256, 0, stream>>>(Wv, wvb, nw);
    f32_to_bf16<<<(nw + 255) / 256, 256, 0, stream>>>(Wo, wob, nw);
  }

  // 2) Q/K/V projections: (BN x D) @ (D x D)ᵀ, 32x64 tile per wave, 8 waves/block
  int gemm_blocks = (kBN / 32) * (kD / 64) / 8;  // 256
  gemm_bf16<1><<<gemm_blocks, 256, 0, stream>>>(xb, wqb, (void*)Qb, kBN, kD, kD);
  gemm_bf16<1><<<gemm_blocks, 256, 0, stream>>>(xb, wkb, (void*)Kb, kBN, kD, kD);
  gemm_bf16<2><<<gemm_blocks, 256, 0, stream>>>(xb, wvb, (void*)Vt, kBN, kD, kD);

  // 3) row norms
  {
    int rows = kBH * kN;
    row_norms<<<(rows + 255) / 256, 256, 0, stream>>>(Qb, Kb, qn, kn, rows);
  }

  // 4) causal hyperbolic attention: 4096 waves, 8 per block
  hyp_attn<<<(kBH * (kN / 16)) / 8, 256, 0, stream>>>(Qb, Kb, Vt, qn, kn, log_c,
                                                      log_beta, Ob);

  // 5) output projection straight to d_out (f32)
  gemm_bf16<0><<<gemm_blocks, 256, 0, stream>>>(Ob, wob, d_out, kBN, kD, kD);
}
